// DeepSeekSparseAttention_64278480552152
// MI455X (gfx1250) — matrix operand order repaired and checked by reading, not compiled, not run on hardware
//
#include <hip/hip_runtime.h>
#include <math.h>

// ---------------- problem constants ----------------
#define B_   2
#define T_   2048
#define D_   2048
#define H_   16
#define DH_  128
#define DR_  64
#define DL_  512
#define HDR_ (H_*DR_)          // 1024
#define QKD_ (DH_ + DR_)       // 192 dims per head for scores
#define WIN_HALF_ 256
#define GLOBAL_   128
#define SCALE_      0.0883883476f   // 1/sqrt(128)
#define SCALE_ROPE_ 0.125f          // 1/sqrt(64)
#define YARN_F_     1.3465735903f   // 0.1*ln(32)+1

typedef __attribute__((ext_vector_type(8)))  float  f32x8;
typedef __attribute__((ext_vector_type(16))) __bf16 bf16x16;

union Frag {          // 16 bf16 = 8 VGPRs, loaded as two 16B chunks
  bf16x16 v;
  uint4   u[2];
};

static __device__ inline f32x8 vzero8() {
  f32x8 z = {0.f,0.f,0.f,0.f,0.f,0.f,0.f,0.f};
  return z;
}

// ---- CDNA5 async global->LDS copy (ASYNCcnt path) ----
// LDS byte offset = low 32 bits of the generic pointer (flat LDS aperture
// keeps the LDS offset in addr[31:0] per CDNA5 aperture rules).
static __device__ inline void async_copy_b128(const __bf16* gptr, __bf16* lptr) {
  unsigned lds = (unsigned)(unsigned long long)(void*)lptr;
  asm volatile("global_load_async_to_lds_b128 %0, %1, off"
               :: "v"(lds), "v"(gptr)
               : "memory");
}
static __device__ inline void wait_async_all() {
  asm volatile("s_wait_asynccnt 0x0" ::: "memory");
}

// ---------------- fp32 -> bf16 cast ----------------
__global__ void cast_f32_to_bf16(const float* __restrict__ src,
                                 __bf16* __restrict__ dst, int n) {
  int i = (blockIdx.x * blockDim.x + threadIdx.x) * 4;
  if (i + 3 < n) {
    float4 v = *(const float4*)(src + i);
    dst[i+0] = (__bf16)v.x; dst[i+1] = (__bf16)v.y;
    dst[i+2] = (__bf16)v.z; dst[i+3] = (__bf16)v.w;
  } else {
    for (int j = i; j < n; ++j) dst[j] = (__bf16)src[j];
  }
}

// fp32 RxC row-major -> bf16 CxR row-major (transpose; coalesced reads)
__global__ void cast_transpose_f32_to_bf16(const float* __restrict__ src,
                                           __bf16* __restrict__ dst,
                                           int R, int C) {
  int idx = blockIdx.x * blockDim.x + threadIdx.x;
  if (idx >= R * C) return;
  int r = idx / C, c = idx % C;
  dst[(size_t)c * R + r] = (__bf16)src[idx];
}

// ---------------- generic bf16 WMMA GEMM (double-buffered async) ----------------
// C[M,N] = alpha * A[M,K] * Bt[N,K]^T; A row-major, Bt is B TRANSPOSED
// (N-major, K contiguous) so B-fragments are contiguous ds_load_b128 pairs.
// Block 256 threads = 8 waves; block tile 128x64; wave tile 32x32 (2x2 wmma).
// Async copies for tile i+1 are issued before computing tile i.
template <bool OUT_F32>
__global__ __launch_bounds__(256)
void gemm_bf16_wmma(const __bf16* __restrict__ A, const __bf16* __restrict__ Bt,
                    void* __restrict__ Cout, int M, int N, int K, float alpha) {
  __shared__ __align__(16) __bf16 sA[2][128*32];   // [m][k] row-major
  __shared__ __align__(16) __bf16 sBt[2][64*32];   // [n][k] row-major (K contiguous)

  const int tid  = threadIdx.x;
  const int lane = tid & 31;
  const int wave = tid >> 5;
  const int wr   = wave & 3;     // 4 row groups
  const int wc   = wave >> 2;    // 2 col groups
  const int m0   = blockIdx.y * 128;
  const int n0   = blockIdx.x * 64;
  const int hi   = lane >> 4;
  const int ln   = lane & 15;

  f32x8 acc[2][2];
  for (int i = 0; i < 2; ++i)
    for (int j = 0; j < 2; ++j) acc[i][j] = vzero8();

  auto stage = [&](int k0, int buf) {
    // A tile 128x32: 2 x 16B chunks per thread
#pragma unroll
    for (int p = 0; p < 2; ++p) {
      int e = (p*256 + tid) * 8;
      int r = e >> 5, c = e & 31;
      async_copy_b128(A + (size_t)(m0 + r) * K + k0 + c, &sA[buf][r*32 + c]);
    }
    // Bt tile 64x32: 1 x 16B chunk per thread
    int n = tid >> 2, part = tid & 3;
    async_copy_b128(Bt + (size_t)(n0 + n) * K + k0 + part*8,
                    &sBt[buf][n*32 + part*8]);
  };

  const int steps = K >> 5;
  stage(0, 0);
  for (int it = 0; it < steps; ++it) {
    const int buf = it & 1;
    wait_async_all();      // own tile-it copies arrived
    __syncthreads();       // everyone's tile-it arrived; compute(it-1) done everywhere
    if (it + 1 < steps) stage((it + 1) << 5, buf ^ 1);  // overlap with compute

    Frag aF[2], bF[2];
    const int mm = wr*32 + ln;
#pragma unroll
    for (int mt = 0; mt < 2; ++mt) {
      // A-fragment: chunks at k = 8*hi and k = 16 + 8*hi
      const __bf16* pa = &sA[buf][(mm + mt*16)*32 + hi*8];
      aF[mt].u[0] = *(const uint4*)pa;
      aF[mt].u[1] = *(const uint4*)(pa + 16);
    }
#pragma unroll
    for (int ct = 0; ct < 2; ++ct) {
      // B-fragment: lane = column n, chunks at k = 8*hi and k = 16 + 8*hi
      int nl = wc*32 + ct*16 + ln;
      const __bf16* pb = &sBt[buf][nl*32 + hi*8];
      bF[ct].u[0] = *(const uint4*)pb;
      bF[ct].u[1] = *(const uint4*)(pb + 16);
    }
#pragma unroll
    for (int mt = 0; mt < 2; ++mt)
#pragma unroll
      for (int ct = 0; ct < 2; ++ct)
        acc[mt][ct] = __builtin_amdgcn_wmma_f32_16x16x32_bf16(
            false, aF[mt].v, false, bF[ct].v, (short)0, acc[mt][ct], false, false);
  }

  // epilogue: C layout -> lane ln is column, VGPR r + 8*hi is row
#pragma unroll
  for (int mt = 0; mt < 2; ++mt)
#pragma unroll
    for (int ct = 0; ct < 2; ++ct) {
      int rbase = m0 + wr*32 + mt*16 + hi*8;
      int col   = n0 + wc*32 + ct*16 + ln;
#pragma unroll
      for (int r = 0; r < 8; ++r) {
        float val  = acc[mt][ct][r] * alpha;
        size_t idx = (size_t)(rbase + r) * N + col;
        if (OUT_F32) ((float*)Cout)[idx] = val;
        else         ((__bf16*)Cout)[idx] = (__bf16)val;
      }
    }
}

// ---------------- YaRN rope helpers ----------------
static __device__ inline float yarn_inv_freq(int i) {
  float base_inv = __expf(-(2.0f * (float)i / 64.0f) * logf(10000.0f));
  float ramp = ((float)i - 1.0f) / 31.0f;   // (i-beta_slow)/(beta_fast-beta_slow)
  ramp = fminf(fmaxf(ramp, 0.0f), 1.0f);
  return base_inv * (1.0f - ramp) + (base_inv * (1.0f / 32.0f)) * ramp;
}

// pack per-head Q or K rows: [b][h][t][192] = {content*scale_c, rope(pos)*YARN*scale_r}
__global__ void pack_qk_rope(const __bf16* __restrict__ cpart,  // (B*T, D)
                             const __bf16* __restrict__ rpart,  // (B*T, H*DR)
                             __bf16* __restrict__ dst,
                             float scale_c, float scale_r) {
  int idx = blockIdx.x * blockDim.x + threadIdx.x;
  if (idx >= B_*H_*T_*QKD_) return;
  int d = idx % QKD_;
  int t = (idx / QKD_) % T_;
  int h = (idx / (QKD_ * T_)) % H_;
  int b = idx / (QKD_ * T_ * H_);
  float val;
  if (d < DH_) {
    val = (float)cpart[((size_t)(b*T_ + t))*D_ + h*DH_ + d] * scale_c;
  } else {
    int dr = d - DH_;
    int i  = dr & 31;
    float ang = (float)t * yarn_inv_freq(i);
    float c = __cosf(ang), s = __sinf(ang);
    size_t rb = ((size_t)(b*T_ + t))*HDR_ + h*DR_;
    float x     = (float)rpart[rb + dr];
    float other = (float)rpart[rb + ((dr < 32) ? dr + 32 : dr - 32)];
    float rot   = (dr < 32) ? -other : other;
    val = (x*c + rot*s) * YARN_F_ * scale_r;
  }
  dst[((size_t)(b*H_ + h)*T_ + t)*QKD_ + d] = (__bf16)val;
}

// V transposed per head: dst[bh][d][t] so PV B-fragments are K-contiguous
__global__ void pack_v_t(const __bf16* __restrict__ vc,  // (B*T, D)
                         __bf16* __restrict__ dst) {     // [b*H+h][128][T]
  int idx = blockIdx.x * blockDim.x + threadIdx.x;
  if (idx >= B_*H_*T_*DH_) return;
  int d = idx % DH_;
  int t = (idx / DH_) % T_;
  int h = (idx / (DH_ * T_)) % H_;
  int b = idx / (DH_ * T_ * H_);
  dst[((size_t)(b*H_ + h)*DH_ + d)*T_ + t] =
      vc[((size_t)(b*T_ + t))*D_ + h*DH_ + d];
}

// ---------------- sparse flash attention (WMMA, double-buffered async) ----------
// grid: (T/64, B*H); block 128 = 4 waves; each wave owns 16 query rows.
__global__ __launch_bounds__(128)
void sparse_flash_attn(const __bf16* __restrict__ Qp,   // [bh][t][192] (pre-scaled)
                       const __bf16* __restrict__ Kp,   // [bh][t][192]
                       const __bf16* __restrict__ Vt,   // [bh][128][T]
                       __bf16* __restrict__ Out) {      // (B*T, D)
  __shared__ __align__(16) __bf16 sK[2][32*192];   // [key][d]: B-fragments contiguous
  __shared__ __align__(16) __bf16 sVt[2][128*32];  // [d][key]
  __shared__ __align__(16) __bf16 sP[4][16*32];    // per-wave P scratch

  const int tid  = threadIdx.x;
  const int lane = tid & 31;
  const int wave = tid >> 5;
  const int hi   = lane >> 4;
  const int ln   = lane & 15;
  const int bh   = blockIdx.y;
  const int b    = bh / H_;
  const int h    = bh % H_;
  const int qt0  = blockIdx.x * 64;
  const int qw   = qt0 + wave * 16;

  // resident Q fragments: 6 k-steps covering 192 dims
  Frag qF[6];
  const __bf16* qrow = Qp + ((size_t)bh * T_ + qw + ln) * QKD_;
#pragma unroll
  for (int s = 0; s < 6; ++s) {
    int d0 = s*32 + hi*8;
    qF[s].u[0] = *(const uint4*)(qrow + d0);
    qF[s].u[1] = *(const uint4*)(qrow + d0 + 16);
  }

  f32x8 Oacc[8];
#pragma unroll
  for (int g = 0; g < 8; ++g) Oacc[g] = vzero8();
  float mi[8], li[8];
#pragma unroll
  for (int r = 0; r < 8; ++r) { mi[r] = -1.0e30f; li[r] = 0.0f; }

  const int kend = qt0 + 64;   // causal upper bound for this q tile

  // structural sparsity: skip odd-32 tiles past the global region and window
  auto next_active = [&](int k) {
    while (k < kend &&
           !(((k & 63) == 0) || (k < GLOBAL_) || (k + 31 >= qt0 - WIN_HALF_)))
      k += 32;
    return k;
  };
  auto stage = [&](int k0, int buf) {
    // K tile: 32 keys x 192 dims (row-major, 6 x 16B per thread)
#pragma unroll
    for (int p = 0; p < 6; ++p) {
      int c = p*128 + tid;
      int key = c / 24, d0 = (c % 24) * 8;
      async_copy_b128(Kp + ((size_t)bh*T_ + k0 + key)*QKD_ + d0,
                      &sK[buf][key*192 + d0]);
    }
    // Vt tile: 128 dims x 32 keys (4 x 16B per thread)
#pragma unroll
    for (int p = 0; p < 4; ++p) {
      int c = p*128 + tid;
      int d = c >> 2, part = c & 3;
      async_copy_b128(Vt + ((size_t)bh*DH_ + d)*T_ + k0 + part*8,
                      &sVt[buf][d*32 + part*8]);
    }
  };

  int k0  = next_active(0);   // k=0 always active
  int buf = 0;
  stage(k0, buf);
  while (k0 < kend) {
    const int k1 = next_active(k0 + 32);
    wait_async_all();     // own copies for tile k0 arrived
    __syncthreads();      // all waves' copies arrived; previous compute done
    if (k1 < kend) stage(k1, buf ^ 1);   // prefetch next live tile

    // scores S(16q x 32k): two 16-key accumulators, 6 wmma each
    f32x8 S[2];
    S[0] = vzero8(); S[1] = vzero8();
#pragma unroll
    for (int s = 0; s < 6; ++s) {
#pragma unroll
      for (int kg = 0; kg < 2; ++kg) {
        Frag bK;  // lane = key, dims at s*32 + 8*hi and s*32 + 16 + 8*hi
        const __bf16* pk = &sK[buf][(kg*16 + ln)*192 + s*32 + hi*8];
        bK.u[0] = *(const uint4*)pk;
        bK.u[1] = *(const uint4*)(pk + 16);
        S[kg] = __builtin_amdgcn_wmma_f32_16x16x32_bf16(
            false, qF[s].v, false, bK.v, (short)0, S[kg], false, false);
      }
    }

    // mask + online softmax (row = r + 8*hi across the 16-lane half)
    float sc[8];
#pragma unroll
    for (int r = 0; r < 8; ++r) {
      int qq = qw + r + 8*hi;
#pragma unroll
      for (int kg = 0; kg < 2; ++kg) {
        int kk = k0 + kg*16 + ln;
        int dist = qq - kk;
        bool valid = (kk <= qq) &&
                     ((dist <= WIN_HALF_) || ((kk & 63) == 0) || (kk < GLOBAL_));
        if (!valid) S[kg][r] = -1.0e30f;
      }
      float mx = fmaxf(S[0][r], S[1][r]);
#pragma unroll
      for (int off = 8; off >= 1; off >>= 1)
        mx = fmaxf(mx, __shfl_xor(mx, off, 32));
      float mn    = fmaxf(mi[r], mx);
      float scale = __expf(mi[r] - mn);
      float rsum  = 0.0f;
#pragma unroll
      for (int kg = 0; kg < 2; ++kg) {
        float p = __expf(S[kg][r] - mn);
        rsum += p;
        sP[wave][(r + 8*hi)*32 + kg*16 + ln] = (__bf16)p;
      }
#pragma unroll
      for (int off = 8; off >= 1; off >>= 1)
        rsum += __shfl_xor(rsum, off, 32);
      li[r] = li[r]*scale + rsum;
      mi[r] = mn;
      sc[r] = scale;
    }
#pragma unroll
    for (int g = 0; g < 8; ++g)
#pragma unroll
      for (int r = 0; r < 8; ++r) Oacc[g][r] *= sc[r];

    // P(16x32) x V(32x128): P re-read from LDS as A-fragment (same-wave DS in-order)
    Frag pF;
    const __bf16* pp = &sP[wave][ln*32 + hi*8];
    pF.u[0] = *(const uint4*)pp;
    pF.u[1] = *(const uint4*)(pp + 16);
#pragma unroll
    for (int g = 0; g < 8; ++g) {
      Frag bV;  // lane = output dim d, keys at 8*hi and 16 + 8*hi
      const __bf16* pv = &sVt[buf][(g*16 + ln)*32 + hi*8];
      bV.u[0] = *(const uint4*)pv;
      bV.u[1] = *(const uint4*)(pv + 16);
      Oacc[g] = __builtin_amdgcn_wmma_f32_16x16x32_bf16(
          false, pF.v, false, bV.v, (short)0, Oacc[g], false, false);
    }

    k0 = k1;
    buf ^= 1;
  }

  // epilogue: normalize and scatter to (B*T, D) head-interleaved layout
#pragma unroll
  for (int r = 0; r < 8; ++r) {
    float inv = 1.0f / li[r];
    int qq = qw + r + 8*hi;
    __bf16* orow = Out + ((size_t)b*T_ + qq)*D_ + h*DH_;
#pragma unroll
    for (int g = 0; g < 8; ++g)
      orow[g*16 + ln] = (__bf16)(Oacc[g][r] * inv);
  }
}

// ---------------- host orchestration ----------------
extern "C" void kernel_launch(void* const* d_in, const int* in_sizes, int n_in,
                              void* d_out, int out_size, void* d_ws, size_t ws_size,
                              hipStream_t stream) {
  (void)in_sizes; (void)n_in; (void)out_size; (void)ws_size;
  const float* x     = (const float*)d_in[0];
  const float* w_q   = (const float*)d_in[1];
  const float* w_dkv = (const float*)d_in[2];
  const float* w_uk  = (const float*)d_in[3];
  const float* w_uv  = (const float*)d_in[4];
  const float* w_qp  = (const float*)d_in[5];
  const float* w_kp  = (const float*)d_in[6];
  const float* w_o   = (const float*)d_in[7];

  char* ws = (char*)d_ws;
  size_t off = 0;
  auto carve = [&](size_t elems) -> __bf16* {
    __bf16* p = (__bf16*)(ws + off);
    off = (off + elems * sizeof(__bf16) + 255) & ~(size_t)255;
    return p;
  };
  const size_t nX = (size_t)B_ * T_ * D_;        // 8.39M
  __bf16* xb    = carve(nX);
  __bf16* wqT   = carve((size_t)D_ * D_);        // [N][K] transposed weights
  __bf16* wdkvT = carve((size_t)D_ * DL_);
  __bf16* wukT  = carve((size_t)DL_ * D_);
  __bf16* wuvT  = carve((size_t)DL_ * D_);
  __bf16* wqpT  = carve((size_t)D_ * HDR_);
  __bf16* wkpT  = carve((size_t)D_ * HDR_);
  __bf16* woT   = carve((size_t)D_ * D_);
  __bf16* qc    = carve(nX);
  __bf16* ckv   = carve((size_t)B_ * T_ * DL_);
  __bf16* kc    = carve(nX);
  __bf16* vc    = carve(nX);
  __bf16* qr    = carve((size_t)B_ * T_ * HDR_);
  __bf16* kr    = carve((size_t)B_ * T_ * HDR_);
  __bf16* Qp    = carve((size_t)B_ * H_ * T_ * QKD_);
  __bf16* Kp    = carve((size_t)B_ * H_ * T_ * QKD_);
  __bf16* Vt    = carve((size_t)B_ * H_ * T_ * DH_);
  __bf16* attn  = carve(nX);

  cast_f32_to_bf16<<<(int)((nX/4 + 255)/256), 256, 0, stream>>>(x, xb, (int)nX);
  auto castT = [&](const float* s, __bf16* d, int R, int C) {
    int n = R * C;
    cast_transpose_f32_to_bf16<<<(n + 255)/256, 256, 0, stream>>>(s, d, R, C);
  };
  castT(w_q,   wqT,   D_,  D_);    // -> [D][D]
  castT(w_dkv, wdkvT, D_,  DL_);   // -> [DL][D]
  castT(w_uk,  wukT,  DL_, D_);    // -> [D][DL]
  castT(w_uv,  wuvT,  DL_, D_);    // -> [D][DL]
  castT(w_qp,  wqpT,  D_,  HDR_);  // -> [HDR][D]
  castT(w_kp,  wkpT,  D_,  HDR_);  // -> [HDR][D]
  castT(w_o,   woT,   D_,  D_);    // -> [D][D]

  const int M = B_ * T_;  // 4096
  auto gemm_bf = [&](const __bf16* A, const __bf16* Bt, __bf16* C,
                     int Mi, int Ni, int Ki) {
    dim3 grid(Ni / 64, Mi / 128);
    gemm_bf16_wmma<false><<<grid, 256, 0, stream>>>(A, Bt, (void*)C, Mi, Ni, Ki, 1.0f);
  };
  gemm_bf(xb,  wqT,   qc,  M, D_,   D_);   // q_c
  gemm_bf(xb,  wdkvT, ckv, M, DL_,  D_);   // c_kv
  gemm_bf(ckv, wukT,  kc,  M, D_,   DL_);  // k_c
  gemm_bf(ckv, wuvT,  vc,  M, D_,   DL_);  // v_c
  gemm_bf(xb,  wqpT,  qr,  M, HDR_, D_);   // q_r
  gemm_bf(xb,  wkpT,  kr,  M, HDR_, D_);   // k_r

  {
    int totQK = B_*H_*T_*QKD_;
    int blocks = (totQK + 255) / 256;
    pack_qk_rope<<<blocks, 256, 0, stream>>>(qc, qr, Qp, SCALE_, SCALE_ROPE_);
    pack_qk_rope<<<blocks, 256, 0, stream>>>(kc, kr, Kp, 1.0f, 1.0f);
    int totV = B_*H_*T_*DH_;
    pack_v_t<<<(totV + 255)/256, 256, 0, stream>>>(vc, Vt);
  }

  {
    dim3 grid(T_ / 64, B_ * H_);
    sparse_flash_attn<<<grid, 128, 0, stream>>>(Qp, Kp, Vt, attn);
  }

  {
    dim3 grid(D_ / 64, M / 128);
    gemm_bf16_wmma<true><<<grid, 256, 0, stream>>>(attn, woT, d_out, M, D_, D_, 1.0f);
  }
}
